// GraphMemoryVQ_24902220382603
// MI455X (gfx1250) — matrix operand order repaired and checked by reading, not compiled, not run on hardware
//
#include <hip/hip_runtime.h>

// ---------------------------------------------------------------------------
// GraphMemoryVQ on gfx1250: fused bf16-WMMA distance GEMM + argmin + gather
// B=16, S=1024, D=512 (2D=1024), K=4096
// Blocking: 64 rows/block (4 C-tiles per wave share each B fragment -> 4x
// less L2 codebook traffic), 8 waves split the 4096 codes.
// K-loop kept ROLLED (unroll(disable)) to stop LICM hoisting A fragments;
// B ping-pongs between two register buffers (no copy, no full drains).
// ---------------------------------------------------------------------------

#define BS      16384        // B*S rows
#define D2      1024         // 2*D (feature dim / GEMM K)
#define KC      4096         // number of codes (GEMM N)
#define ROWS_PER_BLOCK 64
#define TILES   4            // 16-row C tiles per wave
#define NBLOCKS (BS / ROWS_PER_BLOCK)   // 256
#define ZB_STRIDE 1032       // padded bf16 row stride (halves) to dodge bank conflicts

// d_out layout: [0, 16777216) z_complex interleaved floats,
//               [16777216]    loss,
//               [16777217, +16384) min_indices (as float)
#define OUT_LOSS_OFF 16777216u
#define OUT_IDX_OFF  16777217u

typedef __attribute__((ext_vector_type(16))) __bf16          v16bf;
typedef __attribute__((ext_vector_type(16))) unsigned short  v16u;
typedef __attribute__((ext_vector_type(8)))  unsigned short  v8u;
typedef __attribute__((ext_vector_type(8)))  float           v8f;

__device__ __forceinline__ unsigned short f2bf(float f) {
    unsigned int u = __float_as_uint(f);
    unsigned int r = u + 0x7FFFu + ((u >> 16) & 1u);   // round-to-nearest-even
    return (unsigned short)(r >> 16);
}

// A fragment (16x32 bf16, ISA 7.12.2): per-lane runs at +0 and +16 halves
__device__ __forceinline__ v16u ldA(const unsigned short* p) {
    v16u r;
    ((v8u*)&r)[0] = *(const v8u*)(p);
    ((v8u*)&r)[1] = *(const v8u*)(p + 16);
    return r;
}
// B fragment (32x16 bf16): per-lane runs at +0 and +16 halves (two b128 loads)
__device__ __forceinline__ v16u ldB(const unsigned short* p) {
    v16u r;
    ((v8u*)&r)[0] = *(const v8u*)(p);
    ((v8u*)&r)[1] = *(const v8u*)(p + 16);
    return r;
}

__device__ __forceinline__ v8f wmma_bf16(v16u a, v16u b, v8f c) {
    return __builtin_amdgcn_wmma_f32_16x16x32_bf16(
        false, __builtin_bit_cast(v16bf, a),
        false, __builtin_bit_cast(v16bf, b),
        (short)0, c, false, false);
}

// ---------------------------------------------------------------------------
// Kernel 1: codebook f32 -> bf16 (ws) and cnorm[k] = ||codebook[k]||^2
// ---------------------------------------------------------------------------
__global__ void vq_prep(const float* __restrict__ codebook,
                        unsigned short* __restrict__ cbbf,
                        float* __restrict__ cnorm) {
    const int k   = blockIdx.x;
    const int tid = threadIdx.x;
    const float* row = codebook + (size_t)k * D2;
    unsigned short* brow = cbbf + (size_t)k * D2;

    float s = 0.0f;
    for (int d = tid; d < D2; d += 256) {
        float v = row[d];
        s += v * v;
        brow[d] = f2bf(v);
    }
    __shared__ float red[8];
    for (int off = 16; off; off >>= 1) s += __shfl_xor(s, off, 32);
    if ((tid & 31) == 0) red[tid >> 5] = s;
    __syncthreads();
    if (tid == 0) {
        float t = 0.0f;
        for (int w = 0; w < 8; ++w) t += red[w];   // fixed order -> deterministic
        cnorm[k] = t;
    }
}

// ---------------------------------------------------------------------------
// Kernel 2: per 64-row tile: z_flat build, WMMA distance sweep (4 tiles/wave),
//           argmin, codebook gather, complex output, loss partial.
// ---------------------------------------------------------------------------
__global__ void __launch_bounds__(256)
vq_main(const float* __restrict__ z_real,
        const float* __restrict__ z_imag,
        const int*   __restrict__ prev_idx,
        const float* __restrict__ offs,
        const float* __restrict__ codebook,
        const float* __restrict__ adj,
        const unsigned short* __restrict__ cbbf,
        const float* __restrict__ cnorm,
        float* __restrict__ partials,
        float* __restrict__ out) {
    extern __shared__ __align__(16) char smem[];
    unsigned short* zbf   = (unsigned short*)smem;                // 64*1032 bf16 (132096 B)
    int*            prevL = (int*)(smem + 132096);                // 64
    int*            chosen= prevL + 64;                           // 64
    float*          wbv   = (float*)(chosen + 64);                // 8*64
    int*            wbi   = (int*)(wbv + 8 * 64);                 // 8*64
    float*          red   = (float*)(wbi + 8 * 64);               // 8

    const int tid  = threadIdx.x;
    const int wv   = tid >> 5;
    const int lane = tid & 31;
    const int m0   = blockIdx.x * ROWS_PER_BLOCK;

    // ---- Stage 1: z_flat = concat(z_real, z_imag) + offset -> LDS bf16
    if (tid < ROWS_PER_BLOCK) prevL[tid] = prev_idx[m0 + tid];
    for (int i = tid; i < ROWS_PER_BLOCK * D2; i += 256) {
        int row = i >> 10;
        int d   = i & (D2 - 1);
        size_t r = (size_t)(m0 + row);
        float z = (d < 512 ? z_real[r * 512 + d] : z_imag[r * 512 + (d - 512)])
                + offs[r * D2 + d];
        zbf[row * ZB_STRIDE + d] = f2bf(z);
    }
    __syncthreads();

    // ---- Stage 2: WMMA distance sweep + running argmin
    const int Ml    = lane & 15;
    const int koffA = (lane >> 4) << 3;   // A: lanes 16-31 shift K runs by +8
    const int koffB = (lane >> 4) << 3;   // B: lanes 16-31 shift K runs by +8
    const unsigned short* abase = zbf + Ml * ZB_STRIDE + koffA;

    // adjacency row offsets in 32-bit elements (prev*4096 fits in 24 bits)
    unsigned adjBase[ROWS_PER_BLOCK / 8];  // only rows this lane-half touches
    // (computed inline below; prevL reads are LDS broadcasts)

    float bestv[TILES][8];
    int   besti[TILES][8];
#pragma unroll
    for (int t = 0; t < TILES; ++t)
#pragma unroll
        for (int v = 0; v < 8; ++v) { bestv[t][v] = 3.4e38f; besti[t][v] = 0; }

#pragma clang loop unroll(disable)
    for (int iter = 0; iter < 32; ++iter) {
        const int ncol = iter * 128 + wv * 16 + Ml;
        const unsigned short* brow = cbbf + (size_t)ncol * D2 + koffB;

        v8f acc[TILES];
#pragma unroll
        for (int t = 0; t < TILES; ++t) acc[t] = (v8f){};

        // Two K-steps per rolled iteration; B ping-pongs between bu/b2 so no
        // register copies and no full-drain waits. Rolled loop keeps ldA
        // addresses loop-carried -> LICM cannot hoist A (spill hazard).
        v16u bu = ldB(brow);                       // k = 0
#pragma clang loop unroll(disable)
        for (int k0 = 0; k0 < D2 - 64; k0 += 64) { // k0 = 0..896
            v16u b2 = ldB(brow + k0 + 32);
#pragma unroll
            for (int t = 0; t < TILES; ++t) {
                v16u au = ldA(abase + (size_t)t * 16 * ZB_STRIDE + k0);
                acc[t] = wmma_bf16(au, bu, acc[t]);
            }
            bu = ldB(brow + k0 + 64);
#pragma unroll
            for (int t = 0; t < TILES; ++t) {
                v16u au = ldA(abase + (size_t)t * 16 * ZB_STRIDE + k0 + 32);
                acc[t] = wmma_bf16(au, b2, acc[t]);
            }
        }
        {   // tail: k0 = 960 (bu already loaded) and 992
            v16u b2 = ldB(brow + (D2 - 32));
#pragma unroll
            for (int t = 0; t < TILES; ++t) {
                v16u au = ldA(abase + (size_t)t * 16 * ZB_STRIDE + (D2 - 64));
                acc[t] = wmma_bf16(au, bu, acc[t]);
            }
#pragma unroll
            for (int t = 0; t < TILES; ++t) {
                v16u au = ldA(abase + (size_t)t * 16 * ZB_STRIDE + (D2 - 32));
                acc[t] = wmma_bf16(au, b2, acc[t]);
            }
        }

        const float cn = cnorm[ncol];
#pragma unroll
        for (int t = 0; t < TILES; ++t) {
#pragma unroll
            for (int v = 0; v < 8; ++v) {
                int M = t * 16 + v + koffA;          // row within block
                // 32-bit index -> SADDR + vgpr-offset addressing, no u64 adds
                unsigned aoff = ((unsigned)prevL[M] << 12) + (unsigned)ncol;
                float g   = adj[aoff];
                // fast sigmoid: v_exp_f32 + v_rcp_f32 (plenty for argmin bias)
                float sig = __builtin_amdgcn_rcpf(1.0f + __expf(-g));
                float dist = cn - 2.0f * acc[t][v] - 0.8f * sig;
                if (dist < bestv[t][v]) { bestv[t][v] = dist; besti[t][v] = ncol; }
            }
        }
    }

    // cross-lane argmin within each 16-lane half (masks stay inside the half)
#pragma unroll
    for (int t = 0; t < TILES; ++t) {
#pragma unroll
        for (int v = 0; v < 8; ++v) {
            float bv = bestv[t][v]; int bi = besti[t][v];
            for (int off = 8; off; off >>= 1) {
                float ov = __shfl_xor(bv, off, 32);
                int   oi = __shfl_xor(bi, off, 32);
                if (ov < bv || (ov == bv && oi < bi)) { bv = ov; bi = oi; }
            }
            if (lane == 0)  { wbv[wv * 64 + t * 16 + v]     = bv; wbi[wv * 64 + t * 16 + v]     = bi; }
            if (lane == 16) { wbv[wv * 64 + t * 16 + v + 8] = bv; wbi[wv * 64 + t * 16 + v + 8] = bi; }
        }
    }
    __syncthreads();

    if (tid < ROWS_PER_BLOCK) {
        float bv = wbv[tid]; int bi = wbi[tid];
        for (int w = 1; w < 8; ++w) {
            float ov = wbv[w * 64 + tid]; int oi = wbi[w * 64 + tid];
            if (ov < bv || (ov == bv && oi < bi)) { bv = ov; bi = oi; }
        }
        chosen[tid] = bi;
        out[OUT_IDX_OFF + (unsigned)(m0 + tid)] = (float)bi;
    }
    __syncthreads();

    // ---- Stage 3: z_q gather, complex output, loss partial
    // (z_flat recomputed from global in f32 for an exact loss)
    float lsum = 0.0f;
    for (int row = 0; row < ROWS_PER_BLOCK; ++row) {
        const int idx = chosen[row];
        const float* cq = codebook + (size_t)idx * D2;
        size_t r = (size_t)(m0 + row);
        float* orow = out + r * D2;                    // 1024 floats per row
        for (int d = tid; d < D2; d += 256) {
            float z = (d < 512 ? z_real[r * 512 + d] : z_imag[r * 512 + (d - 512)])
                    + offs[r * D2 + d];
            float q  = cq[d];
            float df = q - z;
            lsum += df * df;
            if (d < 512) orow[2 * d]             = q;  // real part
            else         orow[2 * (d - 512) + 1] = q;  // imag part
        }
    }
    for (int off = 16; off; off >>= 1) lsum += __shfl_xor(lsum, off, 32);
    if (lane == 0) red[wv] = lsum;
    __syncthreads();
    if (tid == 0) {
        float t = 0.0f;
        for (int w = 0; w < 8; ++w) t += red[w];       // fixed order
        partials[blockIdx.x] = t;
    }
}

// ---------------------------------------------------------------------------
// Kernel 3: deterministic final loss reduction
// ---------------------------------------------------------------------------
__global__ void vq_final(const float* __restrict__ partials, float* __restrict__ out) {
    if (threadIdx.x == 0 && blockIdx.x == 0) {
        float s = 0.0f;
        for (int i = 0; i < NBLOCKS; ++i) s += partials[i];
        // loss_vq + 0.01*loss_commit, both mean((z_q - z_flat)^2) in fwd
        out[OUT_LOSS_OFF] = 1.01f * s / (float)(BS * D2);
    }
}

extern "C" void kernel_launch(void* const* d_in, const int* in_sizes, int n_in,
                              void* d_out, int out_size, void* d_ws, size_t ws_size,
                              hipStream_t stream) {
    const float* z_real = (const float*)d_in[0];
    const float* z_imag = (const float*)d_in[1];
    const int*   prev   = (const int*)  d_in[2];
    const float* offs   = (const float*)d_in[3];
    const float* cb     = (const float*)d_in[4];
    const float* adj    = (const float*)d_in[5];

    unsigned short* cbbf     = (unsigned short*)d_ws;                       // 8 MB
    float*          cnorm    = (float*)((char*)d_ws + (size_t)KC * D2 * 2); // 16 KB
    float*          partials = cnorm + KC;                                  // 1 KB
    float*          out      = (float*)d_out;

    vq_prep<<<KC, 256, 0, stream>>>(cb, cbbf, cnorm);

    const size_t lds = 132096 + 64 * 4 * 2 + 8 * 64 * 4 * 2 + 32;  // ~137 KB
    vq_main<<<NBLOCKS, 256, lds, stream>>>(z_real, z_imag, prev, offs,
                                           cb, adj, cbbf, cnorm, partials, out);

    vq_final<<<1, 32, 0, stream>>>(partials, out);
}
